// CausalSelfAttention_23046794510819
// MI455X (gfx1250) — matrix operand order repaired and checked by reading, not compiled, not run on hardware
//
#include <hip/hip_runtime.h>
#include <hip/hip_bf16.h>

typedef __bf16 bf16;
typedef __bf16 bf16x16 __attribute__((ext_vector_type(16)));
typedef float  f32x8   __attribute__((ext_vector_type(8)));

#define DEVINL __device__ __forceinline__

constexpr int Bsz = 4, Tlen = 2048, Cdim = 1024, NH = 16, HD = 64;
constexpr int Mrows = Bsz * Tlen;  // 8192

DEVINL f32x8 wmma_bf16(bf16x16 a, bf16x16 b, f32x8 c) {
  // D = A(16x32) * B(32x16) + C, f32 accumulate
  return __builtin_amdgcn_wmma_f32_16x16x32_bf16(false, a, false, b, (short)0, c,
                                                 false, false);
}

// A-fragment (16x32 bf16): lane holds row m = lane&15; element e in [0,7] holds
// K = 8*half + e, element e in [8,15] holds K = 16 + 8*half + (e-8)  (half = lane>>4).
// Requires rows contiguous along K. Two 16-byte loads per lane.
DEVINL bf16x16 load_fragA(const bf16* p, int row0, int k0, int ld) {
  const int l = threadIdx.x & 31;
  const bf16* base = p + (long)(row0 + (l & 15)) * ld + k0 + 8 * (l >> 4);
  union { bf16x16 v; uint4 u[2]; } r;
  r.u[0] = *(const uint4*)(base);
  r.u[1] = *(const uint4*)(base + 16);
  return r.v;
}

// B-fragment (32x16 bf16): lane holds column n = lane&15; element e in [0,7] holds
// K = 8*half + e, element e in [8,15] holds K = 16 + 8*half + (e-8). Strided scalar loads.
DEVINL bf16x16 load_fragB(const bf16* p, int k0, int n0, int ld) {
  const int l = threadIdx.x & 31;
  const bf16* base = p + (long)(k0 + 8 * (l >> 4)) * ld + n0 + (l & 15);
  bf16x16 v;
#pragma unroll
  for (int e = 0; e < 8; ++e) {
    v[e] = base[(long)e * ld];
    v[8 + e] = base[(long)(16 + e) * ld];
  }
  return v;
}

// ---- CDNA5 async global->LDS copy (ASYNCcnt-tracked), per-lane 16 bytes ----
DEVINL uint32_t lds_off32(const void* p) {
  // ISA 10.2: LDS aperture maps addr[31:0] to the LDS byte offset.
  return (uint32_t)(size_t)p;
}
DEVINL void async_copy_b128(uint32_t lds_off, const void* gptr) {
  asm volatile("global_load_async_to_lds_b128 %0, %1, off"
               :
               : "v"(lds_off), "v"((unsigned long long)(size_t)gptr)
               : "memory");
}
DEVINL void wait_async0() {
  asm volatile("s_wait_asynccnt 0x0" ::: "memory");
}

// ---------------- bf16 conversion kernels ----------------

__global__ void k_cvt(const float* __restrict__ src, bf16* __restrict__ dst, int n) {
  int i = blockIdx.x * blockDim.x + threadIdx.x;
  if (i < n) dst[i] = (bf16)src[i];
}

// dst[c*C + e] = src[e*C + c]  (transpose C x C so B-fragments read contiguously)
__global__ void k_cvt_w(const float* __restrict__ src, bf16* __restrict__ dst) {
  int i = blockIdx.x * blockDim.x + threadIdx.x;
  int c = i >> 10, e = i & (Cdim - 1);
  dst[i] = (bf16)src[e * Cdim + c];
}

// ---------------- WMMA GEMM: O[m,n] = sum_k A[m,k] * Bt[k,n] ----------------
// A: [Mrows, Cdim] bf16 row-major. Bt: [Cdim, Cdim] bf16 (pre-transposed weight).
// mode 0: store bf16 as [B,H,T,D]        (Q, V)
// mode 1: store bf16 as [B,H,D,T]        (K head-transposed for QK^T B-fragments)
// mode 2: store f32  as [Mrows, Cdim]    (final output)
__global__ __launch_bounds__(256) void k_gemm(const bf16* __restrict__ A,
                                              const bf16* __restrict__ Bt,
                                              bf16* __restrict__ outb,
                                              float* __restrict__ outf,
                                              int mode) {
  const int w = threadIdx.x >> 5, l = threadIdx.x & 31;
  const int half = l >> 4, ln = l & 15;
  const int m0 = blockIdx.y * 128 + (w & 3) * 32;  // 4 row groups of 32
  const int n0 = blockIdx.x * 64 + (w >> 2) * 32;  // 2 col groups of 32
  f32x8 c00{}, c01{}, c10{}, c11{};
  for (int k0 = 0; k0 < Cdim; k0 += 32) {
    if (k0 + 32 < Cdim)
      __builtin_prefetch(A + (long)m0 * Cdim + k0 + 32, 0, 1);  // global_prefetch_b8
    bf16x16 a0 = load_fragA(A, m0, k0, Cdim);
    bf16x16 a1 = load_fragA(A, m0 + 16, k0, Cdim);
    bf16x16 b0 = load_fragB(Bt, k0, n0, Cdim);
    bf16x16 b1 = load_fragB(Bt, k0, n0 + 16, Cdim);
    c00 = wmma_bf16(a0, b0, c00);
    c01 = wmma_bf16(a0, b1, c01);
    c10 = wmma_bf16(a1, b0, c10);
    c11 = wmma_bf16(a1, b1, c11);
  }
#pragma unroll
  for (int i = 0; i < 2; ++i)
#pragma unroll
    for (int j = 0; j < 2; ++j) {
      f32x8 cc = i ? (j ? c11 : c10) : (j ? c01 : c00);
#pragma unroll
      for (int r = 0; r < 8; ++r) {
        const int m = m0 + i * 16 + r + 8 * half;  // C/D layout: row = r + 8*half
        const int n = n0 + j * 16 + ln;            //             col = lane&15
        const float v = cc[r];
        if (mode == 2) {
          outf[(long)m * Cdim + n] = v;
        } else {
          const int bb = m >> 11, tt = m & (Tlen - 1);
          const int hh = n >> 6, dd = n & (HD - 1);
          const long idx =
              (mode == 0) ? ((((long)bb * NH + hh) * Tlen + tt) * HD + dd)
                          : ((((long)bb * NH + hh) * HD + dd) * Tlen + tt);
          outb[idx] = (bf16)v;
        }
      }
    }
}

// ---------------- Flash attention (causal), all-WMMA, async-staged K/V ----------------
// Q: [B,H,T,D], Kt: [B,H,D,T], V: [B,H,T,D], Y: [B,T,C] (heads merged), all bf16.
// Block: 256 threads (8 waves); wave w owns 16 query rows; grid (T/128, H, B).
// K/V chunks (32 keys) are staged once per block into LDS with ASYNCcnt-tracked
// async b128 copies, double-buffered so chunk kc+1 streams while kc computes.
__global__ __launch_bounds__(256) void k_attn(const bf16* __restrict__ Q,
                                              const bf16* __restrict__ Kt,
                                              const bf16* __restrict__ V,
                                              bf16* __restrict__ Y) {
  __shared__ __align__(16) bf16 ldsK[2][HD][32];   // [buf][d][key]   8 KB
  __shared__ __align__(16) bf16 ldsV[2][32][HD];   // [buf][key][d]   8 KB
  __shared__ __align__(16) bf16 ldsP[8][16][32];   // per-wave P tile 8 KB
  const int tid = threadIdx.x;
  const int w = tid >> 5, l = tid & 31;
  const int half = l >> 4, ln = l & 15;
  const int qc = blockIdx.x, h = blockIdx.y, b = blockIdx.z;
  const long ho = ((long)b * NH + h) * Tlen * HD;
  const bf16* Qh = Q + ho;
  const bf16* Kh = Kt + ho;  // [HD][Tlen]
  const bf16* Vh = V + ho;   // [Tlen][HD]
  const int q0 = qc * 128 + w * 16;

  // Issue one K (64 rows x 64B) + one V (32 rows x 128B) chunk: 256 lanes x 16B each.
  auto issue_chunk = [&](int kc, int buf) {
    const int kd = tid >> 2, kseg = tid & 3;  // K: d-row, 16B segment
    async_copy_b128(lds_off32(&ldsK[buf][kd][kseg * 8]),
                    Kh + (long)kd * Tlen + kc * 32 + kseg * 8);
    const int vr = tid >> 3, vseg = tid & 7;  // V: key-row, 16B segment
    async_copy_b128(lds_off32(&ldsV[buf][vr][vseg * 8]),
                    Vh + (long)(kc * 32 + vr) * HD + vseg * 8);
  };

  const bf16x16 qa0 = load_fragA(Qh, q0, 0, HD);
  const bf16x16 qa1 = load_fragA(Qh, q0, 32, HD);
  f32x8 o0{}, o1{}, o2{}, o3{};
  float mrow[8], lrow[8];
#pragma unroll
  for (int r = 0; r < 8; ++r) { mrow[r] = -1e30f; lrow[r] = 0.f; }

  bf16* myP = &ldsP[w][0][0];
  const int nchunks = qc * 4 + 4;  // keys up to block's last query row

  issue_chunk(0, 0);
  wait_async0();
  __syncthreads();

  for (int kc = 0; kc < nchunks; ++kc) {
    const int buf = kc & 1;
    if (kc + 1 < nchunks) issue_chunk(kc + 1, buf ^ 1);  // overlap with compute
    const bf16* Kc = &ldsK[buf][0][0];
    const bf16* Vc = &ldsV[buf][0][0];

    // ---- S = Q K^T for a 16x32 score tile (two 16x16 WMMA tiles) ----
    float sf[2][8];
#pragma unroll
    for (int t = 0; t < 2; ++t) {
      bf16x16 kb0 = load_fragB(Kc, 0, t * 16, 32);
      bf16x16 kb1 = load_fragB(Kc, 32, t * 16, 32);
      f32x8 s{};
      s = wmma_bf16(qa0, kb0, s);
      s = wmma_bf16(qa1, kb1, s);
#pragma unroll
      for (int r = 0; r < 8; ++r) {
        const int qg = q0 + r + 8 * half;
        const int kg = kc * 32 + t * 16 + ln;
        sf[t][r] = (kg <= qg) ? s[r] * 0.125f : -1e30f;  // 1/sqrt(64), causal mask
      }
    }
    // ---- online softmax; row stats replicated across the 16 lanes of a row ----
    float p0[8], p1[8], corr[8];
#pragma unroll
    for (int r = 0; r < 8; ++r) {
      float mx = fmaxf(sf[0][r], sf[1][r]);
      mx = fmaxf(mx, __shfl_xor(mx, 1, 32));
      mx = fmaxf(mx, __shfl_xor(mx, 2, 32));
      mx = fmaxf(mx, __shfl_xor(mx, 4, 32));
      mx = fmaxf(mx, __shfl_xor(mx, 8, 32));
      const float mn = fmaxf(mrow[r], mx);
      corr[r] = __expf(mrow[r] - mn);
      mrow[r] = mn;
      p0[r] = __expf(sf[0][r] - mn);
      p1[r] = __expf(sf[1][r] - mn);
      float sm = p0[r] + p1[r];
      sm += __shfl_xor(sm, 1, 32);
      sm += __shfl_xor(sm, 2, 32);
      sm += __shfl_xor(sm, 4, 32);
      sm += __shfl_xor(sm, 8, 32);
      lrow[r] = lrow[r] * corr[r] + sm;
    }
#pragma unroll
    for (int r = 0; r < 8; ++r) {
      o0[r] *= corr[r]; o1[r] *= corr[r]; o2[r] *= corr[r]; o3[r] *= corr[r];
    }
    // ---- P tile: C/D layout -> A layout via per-wave LDS roundtrip ----
#pragma unroll
    for (int r = 0; r < 8; ++r) {
      myP[(r + 8 * half) * 32 + ln] = (bf16)p0[r];
      myP[(r + 8 * half) * 32 + 16 + ln] = (bf16)p1[r];
    }
#if __has_builtin(__builtin_amdgcn_s_wait_dscnt)
    __builtin_amdgcn_s_wait_dscnt(0);
#endif
    __builtin_amdgcn_wave_barrier();
    const bf16x16 pa = load_fragA(myP, 0, 0, 32);
    // ---- O += P V : staged V[key][d] is exactly B-fragment layout (lane = d column) ----
    const bf16x16 vb0 = load_fragB(Vc, 0, 0, HD);
    const bf16x16 vb1 = load_fragB(Vc, 0, 16, HD);
    const bf16x16 vb2 = load_fragB(Vc, 0, 32, HD);
    const bf16x16 vb3 = load_fragB(Vc, 0, 48, HD);
    o0 = wmma_bf16(pa, vb0, o0);
    o1 = wmma_bf16(pa, vb1, o1);
    o2 = wmma_bf16(pa, vb2, o2);
    o3 = wmma_bf16(pa, vb3, o3);
    __builtin_amdgcn_wave_barrier();

    wait_async0();    // next chunk's async copies have landed in LDS
    __syncthreads();  // all waves done reading buf and done copying buf^1
  }
  // ---- epilogue: normalize and write heads-merged bf16 ----
#pragma unroll
  for (int r = 0; r < 8; ++r) {
    const int qg = q0 + r + 8 * half;
    const float inv = 1.0f / lrow[r];
    const long base = ((long)b * Tlen + qg) * Cdim + h * HD;
    Y[base + 0 + ln] = (bf16)(o0[r] * inv);
    Y[base + 16 + ln] = (bf16)(o1[r] * inv);
    Y[base + 32 + ln] = (bf16)(o2[r] * inv);
    Y[base + 48 + ln] = (bf16)(o3[r] * inv);
  }
}

// ---------------- host ----------------

extern "C" void kernel_launch(void* const* d_in, const int* in_sizes, int n_in,
                              void* d_out, int out_size, void* d_ws, size_t ws_size,
                              hipStream_t stream) {
  (void)in_sizes; (void)n_in; (void)out_size; (void)ws_size;
  const float* x  = (const float*)d_in[0];
  const float* Wk = (const float*)d_in[1];
  const float* Wq = (const float*)d_in[2];
  const float* Wv = (const float*)d_in[3];
  const float* Wp = (const float*)d_in[4];
  float* out = (float*)d_out;

  char* ws = (char*)d_ws;
  constexpr size_t XB = (size_t)Mrows * Cdim * sizeof(bf16);  // 16 MiB
  constexpr size_t WB = (size_t)Cdim * Cdim * sizeof(bf16);   //  2 MiB
  bf16* xb  = (bf16*)(ws);
  bf16* wqt = (bf16*)(ws + XB);
  bf16* wkt = (bf16*)(ws + XB + 1 * WB);
  bf16* wvt = (bf16*)(ws + XB + 2 * WB);
  bf16* wpt = (bf16*)(ws + XB + 3 * WB);
  bf16* Qb  = (bf16*)(ws + 1 * XB + 4 * WB);
  bf16* Ktb = (bf16*)(ws + 2 * XB + 4 * WB);
  bf16* Vb  = (bf16*)(ws + 3 * XB + 4 * WB);
  bf16* Yb  = (bf16*)(ws + 4 * XB + 4 * WB);  // total ~88 MiB

  k_cvt<<<(Mrows * Cdim) / 256, 256, 0, stream>>>(x, xb, Mrows * Cdim);
  k_cvt_w<<<(Cdim * Cdim) / 256, 256, 0, stream>>>(Wq, wqt);
  k_cvt_w<<<(Cdim * Cdim) / 256, 256, 0, stream>>>(Wk, wkt);
  k_cvt_w<<<(Cdim * Cdim) / 256, 256, 0, stream>>>(Wv, wvt);
  k_cvt_w<<<(Cdim * Cdim) / 256, 256, 0, stream>>>(Wp, wpt);

  dim3 gg(Cdim / 64, Mrows / 128);
  k_gemm<<<gg, 256, 0, stream>>>(xb, wqt, Qb, nullptr, 0);   // Q  [B,H,T,D]
  k_gemm<<<gg, 256, 0, stream>>>(xb, wkt, Ktb, nullptr, 1);  // K^T [B,H,D,T]
  k_gemm<<<gg, 256, 0, stream>>>(xb, wvt, Vb, nullptr, 0);   // V  [B,H,T,D]

  k_attn<<<dim3(Tlen / 128, NH, Bsz), 256, 0, stream>>>(Qb, Ktb, Vb, Yb);

  k_gemm<<<gg, 256, 0, stream>>>(Yb, wpt, nullptr, out, 2);  // f32 output proj
}
